// WaveManifoldBlock_77378130804844
// MI455X (gfx1250) — compile-verified, matrix operand order checked by reading, not run on hardware
//
#include <hip/hip_runtime.h>
#include <hip/hip_bf16.h>
#include <cstdint>
#include <cstddef>

// ---------------- problem constants ----------------
#define B_   2
#define S_   1024
#define D_   1024
#define DI_  2048
#define KCONV_ 4
#define H_   16
#define DH_  64
#define NST_ 16      // mamba state dim N
#define DTR_ 64      // DT_RANK
#define NBC_ 96      // DT_RANK + 2*N
#define M_   (B_ * S_)   // 2048 tokens

typedef __attribute__((ext_vector_type(16))) _Float16 v16h;
typedef __attribute__((ext_vector_type(8)))  float    v8f;
typedef __attribute__((ext_vector_type(8)))  _Float16 h8;

// ---------------- weight transpose + cast: W[K,N] f32 -> Wt[N,K] f16 ----------------
// LDS-tiled so both the global read and write are coalesced.
__global__ void transpose_f32_to_f16_kernel(const float* __restrict__ src,
                                            _Float16* __restrict__ dst,
                                            int K, int N) {
    __shared__ float tile[32][33];
    int kb = blockIdx.x * 32, nb = blockIdx.y * 32;
    int tx = threadIdx.x & 31, ty = threadIdx.x >> 5;   // 8 waves
    #pragma unroll
    for (int i = ty; i < 32; i += 8) {
        int k = kb + i, n = nb + tx;
        tile[i][tx] = (k < K && n < N) ? src[(size_t)k * N + n] : 0.f;
    }
    __syncthreads();
    #pragma unroll
    for (int i = ty; i < 32; i += 8) {
        int n = nb + i, k = kb + tx;
        if (n < N && k < K) dst[(size_t)n * K + k] = (_Float16)tile[tx][i];
    }
}

// ---------------- layernorm -> f16 ----------------
__global__ void ln_to_f16_kernel(const float* __restrict__ x,
                                 const float* __restrict__ g,
                                 const float* __restrict__ b,
                                 _Float16* __restrict__ out, int Dd) {
    int row = blockIdx.x;
    const float* xr = x + (size_t)row * Dd;
    float s = 0.f, s2 = 0.f;
    for (int i = threadIdx.x; i < Dd; i += blockDim.x) {
        float v = xr[i]; s += v; s2 += v * v;
    }
    #pragma unroll
    for (int off = 16; off > 0; off >>= 1) {
        s  += __shfl_down(s,  off, 32);
        s2 += __shfl_down(s2, off, 32);
    }
    __shared__ float sh0[8], sh1[8];
    int wid = threadIdx.x >> 5;
    if ((threadIdx.x & 31) == 0) { sh0[wid] = s; sh1[wid] = s2; }
    __syncthreads();
    if (threadIdx.x == 0) {
        float ts = 0.f, ts2 = 0.f;
        int nw = blockDim.x >> 5;
        for (int w = 0; w < nw; ++w) { ts += sh0[w]; ts2 += sh1[w]; }
        sh0[0] = ts; sh1[0] = ts2;
    }
    __syncthreads();
    float mean = sh0[0] / (float)Dd;
    float var  = sh1[0] / (float)Dd - mean * mean;
    float inv  = rsqrtf(var + 1e-5f);
    for (int i = threadIdx.x; i < Dd; i += blockDim.x)
        out[(size_t)row * Dd + i] = (_Float16)((xr[i] - mean) * inv * g[i] + b[i]);
}

// ---------------- WMMA fragment loader ----------------
// ISA 16-bit A 16x32 per-lane layout: elements [2v,2v+1] hold
// K = halfk+2v (v<4) and K = 16+halfk+2(v-4) (v>=4)  -> two contiguous
// 8-element (16B) chunks at +halfk and +16+halfk.  Two global_load_b128.
__device__ __forceinline__ v16h load_frag16x32(const _Float16* p, int halfk) {
    h8 lo = *reinterpret_cast<const h8*>(p + halfk);
    h8 hi = *reinterpret_cast<const h8*>(p + 16 + halfk);
    v16h r;
    #pragma unroll
    for (int i = 0; i < 8; ++i) { r[i] = lo[i]; r[i + 8] = hi[i]; }
    return r;
}

// ---------------- WMMA tiled GEMM (software-pipelined) ----------------
// C[M,N] = A[M,K](f16) @ B[K,N] (+bias[N]) (+resid[M,N]); B passed TRANSPOSED as Bt[N,K] f16.
// Each wave computes a 16 x (16*NT) tile.  The K-loop is software-pipelined:
// fragment loads for slab k+1 are issued before the WMMAs of slab k, so the
// vmem queue stays full behind the matrix pipe (partial s_wait_loadcnt, not 0).
template<int NT>
__global__ void wmma_gemm_kernel(const _Float16* __restrict__ A,
                                 const _Float16* __restrict__ Bt,
                                 const float* __restrict__ bias,
                                 const float* __restrict__ resid,
                                 float* __restrict__ outF,
                                 _Float16* __restrict__ outH,
                                 int M, int N, int Kd) {
    int wave = blockIdx.x * (blockDim.x >> 5) + (threadIdx.x >> 5);
    int lane = threadIdx.x & 31;
    int ntiles = N / (16 * NT);
    int row0 = (wave / ntiles) << 4;
    int col0 = (wave % ntiles) * (16 * NT);
    if (row0 >= M) return;   // wave-uniform: EXEC stays all-ones for WMMA

    int lm    = lane & 15;          // A row / B col within tile
    int halfk = (lane >> 4) << 3;   // 0 or 8 (K sub-block select)
    const _Float16* arow = A + (size_t)(row0 + lm) * Kd;
    const _Float16* brow[NT];
    #pragma unroll
    for (int t = 0; t < NT; ++t) brow[t] = Bt + (size_t)(col0 + 16 * t + lm) * Kd;

    v8f c[NT];
    #pragma unroll
    for (int t = 0; t < NT; ++t) c[t] = (v8f){};

    // prologue: fragments for k0 = 0
    v16h a = load_frag16x32(arow, halfk);
    v16h b[NT];
    #pragma unroll
    for (int t = 0; t < NT; ++t) b[t] = load_frag16x32(brow[t], halfk);

    for (int k0 = 32; k0 < Kd; k0 += 32) {
        // issue next slab's loads before consuming current fragments
        if (k0 + 32 < Kd) {   // prefetch one slab further ahead
            __builtin_prefetch(arow + k0 + 32, 0, 1);
            __builtin_prefetch(brow[0] + k0 + 32, 0, 1);
        }
        v16h an = load_frag16x32(arow + k0, halfk);
        v16h bn[NT];
        #pragma unroll
        for (int t = 0; t < NT; ++t) bn[t] = load_frag16x32(brow[t] + k0, halfk);
        #pragma unroll
        for (int t = 0; t < NT; ++t)
            c[t] = __builtin_amdgcn_wmma_f32_16x16x32_f16(
                       false, a, false, b[t], (short)0, c[t], false, false);
        a = an;
        #pragma unroll
        for (int t = 0; t < NT; ++t) b[t] = bn[t];
    }
    // epilogue: last slab
    #pragma unroll
    for (int t = 0; t < NT; ++t)
        c[t] = __builtin_amdgcn_wmma_f32_16x16x32_f16(
                   false, a, false, b[t], (short)0, c[t], false, false);

    // C/D layout: lanes 0-15 hold M = vgpr, lanes 16-31 hold M = vgpr+8
    int rbase = row0 + ((lane >> 4) << 3);
    #pragma unroll
    for (int t = 0; t < NT; ++t) {
        int n = col0 + 16 * t + lm;
        float bv = bias ? bias[n] : 0.f;
        #pragma unroll
        for (int v = 0; v < 8; ++v) {
            size_t idx = (size_t)(rbase + v) * N + n;
            float val = c[t][v] + bv;
            if (resid) val += resid[idx];
            if (outF) outF[idx] = val;
            else      outH[idx] = (_Float16)val;
        }
    }
}

// ---------------- causal depthwise conv (K=4) + SiLU ----------------
// xz16: [M_, 2*DI_] f16 (xp = cols [0,DI), z = cols [DI,2*DI))
__global__ void conv_silu_kernel(const _Float16* __restrict__ xz16,
                                 const float* __restrict__ conv_w,
                                 const float* __restrict__ conv_b,
                                 _Float16* __restrict__ xc16) {
    int idx = blockIdx.x * blockDim.x + threadIdx.x;
    if (idx >= M_ * DI_) return;
    int g  = idx / DI_;
    int di = idx - g * DI_;
    int bb = g / S_;
    int t  = g - bb * S_;
    float acc = conv_b[di];
    #pragma unroll
    for (int j = 0; j < KCONV_; ++j) {
        int ts = t - (KCONV_ - 1) + j;
        if (ts >= 0)
            acc += conv_w[di * KCONV_ + j] *
                   (float)xz16[((size_t)bb * S_ + ts) * (2 * DI_) + di];
    }
    float sl = acc / (1.f + __expf(-acc));
    xc16[idx] = (_Float16)sl;
}

// ---------------- dtr slice cast (dbc[:, :64] -> f16) ----------------
__global__ void dtr_cast_kernel(const float* __restrict__ dbc,
                                _Float16* __restrict__ dtr16) {
    int idx = blockIdx.x * blockDim.x + threadIdx.x;
    if (idx >= M_ * DTR_) return;
    int r = idx / DTR_, k = idx - r * DTR_;
    dtr16[idx] = (_Float16)dbc[(size_t)r * NBC_ + k];
}

// ---------------- softplus(dt_raw + b_dt) in place ----------------
__global__ void softplus_kernel(float* __restrict__ dt,
                                const float* __restrict__ b_dt, int n) {
    int i = blockIdx.x * blockDim.x + threadIdx.x;
    if (i >= n) return;
    float v = dt[i] + b_dt[i % DI_];
    dt[i] = (v > 20.f) ? v : log1pf(__expf(v));
}

// ---------------- selective scan, fused gating epilogue ----------------
// One thread per (batch, channel): 16-state recurrence over S=1024 steps.
// yg16 = (y + xc*D_skip) * silu(z)  written directly in f16 for the W_out GEMM.
__global__ void scan_fused_kernel(const float* __restrict__ dt,
                                  const _Float16* __restrict__ xc16,
                                  const float* __restrict__ dbc,
                                  const float* __restrict__ A_log,
                                  const float* __restrict__ D_skip,
                                  const _Float16* __restrict__ xz16,
                                  _Float16* __restrict__ yg16) {
    int tid = blockIdx.x * blockDim.x + threadIdx.x;
    if (tid >= B_ * DI_) return;
    int bb = tid / DI_, di = tid - bb * DI_;

    float Ar[NST_], h[NST_];
    #pragma unroll
    for (int n = 0; n < NST_; ++n) {
        Ar[n] = -__expf(A_log[(size_t)di * NST_ + n]);
        h[n] = 0.f;
    }
    float dsk = D_skip[di];

    for (int t = 0; t < S_; ++t) {
        size_t g = (size_t)bb * S_ + t;
        float dtv = dt[g * DI_ + di];
        float xcv = (float)xc16[g * DI_ + di];
        const float* bc = dbc + g * NBC_;
        float dx = dtv * xcv;
        float acc = 0.f;
        #pragma unroll
        for (int n = 0; n < NST_; ++n) {
            h[n] = __expf(dtv * Ar[n]) * h[n] + dx * bc[DTR_ + n];
            acc += h[n] * bc[DTR_ + NST_ + n];
        }
        float z  = (float)xz16[g * (2 * DI_) + DI_ + di];
        float sz = z / (1.f + __expf(-z));
        yg16[g * DI_ + di] = (_Float16)((acc + xcv * dsk) * sz);
    }
}

// ---------------- soliton PDE terms -> f16 (pre-scaled by SOL_DT) ----------------
__global__ void soliton_kernel(const float* __restrict__ uu,
                               const float* __restrict__ gates,
                               const float* __restrict__ alpha,
                               _Float16* __restrict__ du16) {
    int idx = blockIdx.x * blockDim.x + threadIdx.x;
    if (idx >= M_ * D_) return;
    int g = idx / D_, d = idx - g * D_;
    int bb = g / S_, t = g - bb * S_;
    int hh = d / DH_;
    const float* ub = uu + (size_t)bb * S_ * D_;
    auto U = [&](int tt) {
        tt = tt < 0 ? 0 : (tt > S_ - 1 ? S_ - 1 : tt);   // edge padding
        return ub[(size_t)tt * D_ + d];
    };
    float u0 = U(t), up1 = U(t + 1), um1 = U(t - 1), up2 = U(t + 2), um2 = U(t - 2);
    float ux   = 0.5f * (up1 - um1);
    float uxx  = up1 - 2.f * u0 + um1;
    float uxxx = 0.5f * (up2 - 2.f * up1 + 2.f * um1 - um2);
    float sg  = __sinf(u0);
    float kdv = -6.f * u0 * ux - uxxx;
    float hj  = (1.f + alpha[hh] * u0) * uxx;
    float du = gates[hh * 3 + 0] * sg + gates[hh * 3 + 1] * kdv + gates[hh * 3 + 2] * hj;
    du16[idx] = (_Float16)(0.1f * du);   // SOL_DT folded in
}

// ---------------- host side ----------------
static inline int ceil_div(int a, int b) { return (a + b - 1) / b; }

template<int NT>
static void launch_gemm(const _Float16* A, const _Float16* Bt, const float* bias,
                        const float* resid, float* outF, _Float16* outH,
                        int M, int N, int Kd, hipStream_t stream) {
    int waves = (M >> 4) * (N / (16 * NT));
    int blocks = ceil_div(waves, 8);
    wmma_gemm_kernel<NT><<<blocks, 256, 0, stream>>>(A, Bt, bias, resid, outF, outH, M, N, Kd);
}

extern "C" void kernel_launch(void* const* d_in, const int* in_sizes, int n_in,
                              void* d_out, int out_size, void* d_ws, size_t ws_size,
                              hipStream_t stream) {
    const float* x      = (const float*)d_in[0];
    const float* ln1_g  = (const float*)d_in[1];
    const float* ln1_b  = (const float*)d_in[2];
    const float* W_in   = (const float*)d_in[3];
    const float* b_in   = (const float*)d_in[4];
    const float* conv_w = (const float*)d_in[5];
    const float* conv_b = (const float*)d_in[6];
    const float* W_xp   = (const float*)d_in[7];
    const float* W_dt   = (const float*)d_in[8];
    const float* b_dt   = (const float*)d_in[9];
    const float* A_log  = (const float*)d_in[10];
    const float* D_skip = (const float*)d_in[11];
    const float* W_out  = (const float*)d_in[12];
    const float* b_out  = (const float*)d_in[13];
    const float* ln2_g  = (const float*)d_in[14];
    const float* ln2_b  = (const float*)d_in[15];
    const float* W_si   = (const float*)d_in[16];
    const float* b_si   = (const float*)d_in[17];
    const float* gates  = (const float*)d_in[18];
    const float* alpha  = (const float*)d_in[19];
    const float* W_so   = (const float*)d_in[20];
    const float* b_so   = (const float*)d_in[21];
    float* out = (float*)d_out;

    // workspace carve-out (256B aligned)
    char* ws = (char*)d_ws;
    size_t off = 0;
    auto carve = [&](size_t bytes) -> void* {
        void* p = ws + off;
        off = (off + bytes + 255) & ~((size_t)255);
        return p;
    };
    _Float16* W_in_t  = (_Float16*)carve((size_t)D_ * 2 * DI_ * 2);   // [4096,1024]
    _Float16* W_xp_t  = (_Float16*)carve((size_t)DI_ * NBC_ * 2);     // [96,2048]
    _Float16* W_dt_t  = (_Float16*)carve((size_t)DTR_ * DI_ * 2);     // [2048,64]
    _Float16* W_out_t = (_Float16*)carve((size_t)DI_ * D_ * 2);       // [1024,2048]
    _Float16* W_si_t  = (_Float16*)carve((size_t)D_ * D_ * 2);        // [1024,1024]
    _Float16* W_so_t  = (_Float16*)carve((size_t)D_ * D_ * 2);        // [1024,1024]
    _Float16* u16     = (_Float16*)carve((size_t)M_ * D_ * 2);
    _Float16* xz16    = (_Float16*)carve((size_t)M_ * 2 * DI_ * 2);
    _Float16* xc16    = (_Float16*)carve((size_t)M_ * DI_ * 2);
    float*    dbc     = (float*)   carve((size_t)M_ * NBC_ * 4);
    _Float16* dtr16   = (_Float16*)carve((size_t)M_ * DTR_ * 2);
    float*    dt      = (float*)   carve((size_t)M_ * DI_ * 4);
    _Float16* yg16    = (_Float16*)carve((size_t)M_ * DI_ * 2);
    float*    x1      = (float*)   carve((size_t)M_ * D_ * 4);
    _Float16* v16     = (_Float16*)carve((size_t)M_ * D_ * 2);
    float*    uu      = (float*)   carve((size_t)M_ * D_ * 4);
    _Float16* du16    = (_Float16*)carve((size_t)M_ * D_ * 2);
    (void)ws_size; (void)n_in; (void)in_sizes; (void)out_size;

    const int TB = 256;
    auto transpose16 = [&](const float* s, _Float16* d, int K, int N) {
        dim3 grid(ceil_div(K, 32), ceil_div(N, 32));
        transpose_f32_to_f16_kernel<<<grid, TB, 0, stream>>>(s, d, K, N);
    };

    // 1) weight transpose+cast: W[K,N] f32 -> Wt[N,K] f16 (contiguous-K WMMA B operand)
    transpose16(W_in,  W_in_t,  D_,   2 * DI_);
    transpose16(W_xp,  W_xp_t,  DI_,  NBC_);
    transpose16(W_dt,  W_dt_t,  DTR_, DI_);
    transpose16(W_out, W_out_t, DI_,  D_);
    transpose16(W_si,  W_si_t,  D_,   D_);
    transpose16(W_so,  W_so_t,  D_,   D_);

    // 2) LN1 -> f16
    ln_to_f16_kernel<<<M_, TB, 0, stream>>>(x, ln1_g, ln1_b, u16, D_);

    // 3) xz = u @ W_in + b_in  (f16 out)
    launch_gemm<4>(u16, W_in_t, b_in, nullptr, nullptr, xz16, M_, 2 * DI_, D_, stream);

    // 4) causal depthwise conv + silu -> xc16
    conv_silu_kernel<<<ceil_div(M_ * DI_, TB), TB, 0, stream>>>(xz16, conv_w, conv_b, xc16);

    // 5) dbc = xc @ W_xp  (f32, N=96 -> NT=2)
    launch_gemm<2>(xc16, W_xp_t, nullptr, nullptr, dbc, nullptr, M_, NBC_, DI_, stream);

    // 6) dtr16 = f16(dbc[:, :64]);  dt_raw = dtr @ W_dt;  dt = softplus(dt_raw + b_dt)
    dtr_cast_kernel<<<ceil_div(M_ * DTR_, TB), TB, 0, stream>>>(dbc, dtr16);
    launch_gemm<4>(dtr16, W_dt_t, nullptr, nullptr, dt, nullptr, M_, DI_, DTR_, stream);
    softplus_kernel<<<ceil_div(M_ * DI_, TB), TB, 0, stream>>>(dt, b_dt, M_ * DI_);

    // 7) selective scan + fused (D_skip, silu(z)) gating -> yg16
    scan_fused_kernel<<<ceil_div(B_ * DI_, TB), TB, 0, stream>>>(
        dt, xc16, dbc, A_log, D_skip, xz16, yg16);

    // 8) x1 = x + yg @ W_out + b_out
    launch_gemm<4>(yg16, W_out_t, b_out, x, x1, nullptr, M_, D_, DI_, stream);

    // 9) LN2 -> f16; uu = v @ W_si + b_si
    ln_to_f16_kernel<<<M_, TB, 0, stream>>>(x1, ln2_g, ln2_b, v16, D_);
    launch_gemm<4>(v16, W_si_t, b_si, nullptr, uu, nullptr, M_, D_, D_, stream);

    // 10) soliton PDE terms -> du16 (SOL_DT pre-applied)
    soliton_kernel<<<ceil_div(M_ * D_, TB), TB, 0, stream>>>(uu, gates, alpha, du16);

    // 11) out = x1 + du @ W_so + b_so
    launch_gemm<4>(du16, W_so_t, b_so, x1, out, nullptr, M_, D_, D_, stream);
}
